// GATNN_86586540687412
// MI455X (gfx1250) — matrix operand order repaired and checked by reading, not compiled, not run on hardware
//
#include <hip/hip_runtime.h>
#include <hip/hip_bf16.h>
#include <stdint.h>

#define N_NODES  50000
#define N_EDGES  640000
#define E_TOT    (N_EDGES + N_NODES)
#define N_GRAPHS 256
#define HC       128
#define HEADS    4
#define HID      32
#define VOCAB    100
#define NEG_SLOPE 0.2f
#define BN_EPS    1e-5f

typedef __attribute__((ext_vector_type(16))) _Float16 v16h;
typedef __attribute__((ext_vector_type(8)))  float    v8f;

// ---------------------------------------------------------------------------
// Atom encoder: h[n][c] = sum_f emb[f][x[n][f]][c]
// ---------------------------------------------------------------------------
__global__ __launch_bounds__(128) void k_atom(const int* __restrict__ x,
                                              const float* __restrict__ emb,
                                              float* __restrict__ h) {
  __shared__ int xf[9];
  const int n = blockIdx.x;
  const int c = threadIdx.x;
  if (c < 9) xf[c] = x[n * 9 + c];
  __syncthreads();
  float s = 0.f;
#pragma unroll
  for (int f = 0; f < 9; ++f)
    s += emb[(f * VOCAB + xf[f]) * HC + c];
  h[n * HC + c] = s;
}

// ---------------------------------------------------------------------------
// Weight convert: W[k][n] f32 -> Wt[n][k] f16 (transposed so K contiguous)
// ---------------------------------------------------------------------------
__global__ void k_wt(const float* __restrict__ Wc, const float* __restrict__ Ws,
                     _Float16* __restrict__ Tc, _Float16* __restrict__ Ts) {
  const int i = blockIdx.x * blockDim.x + threadIdx.x;  // 0..16383
  const int k = i >> 7, n = i & 127;
  Tc[n * 128 + k] = (_Float16)Wc[k * 128 + n];
  Ts[n * 128 + k] = (_Float16)Ws[k * 128 + n];
}

// h f32 -> f16 activations for WMMA A-operand
__global__ void k_h2h(const float* __restrict__ in, _Float16* __restrict__ out, int n) {
  for (int i = blockIdx.x * blockDim.x + threadIdx.x; i < n; i += gridDim.x * blockDim.x)
    out[i] = (_Float16)in[i];
}

// ---------------------------------------------------------------------------
// WMMA GEMM: Out[N][128] = A[N][128] (f16) x W[128][128] (Bt = W^T f16) + bias
// 128 threads = 4 waves; each WG does 64 rows x 128 cols; each wave 16x128.
// K loop: 4 chunks of 32 via v_wmma_f32_16x16x32_f16.
// ---------------------------------------------------------------------------
__global__ __launch_bounds__(128) void k_gemm_wmma(
    const _Float16* __restrict__ A, const _Float16* __restrict__ Bt,
    const float* __restrict__ bias, float* __restrict__ Out) {
  __shared__ __align__(16) _Float16 As[64 * 128];
  __shared__ __align__(16) _Float16 Bs[128 * 128];
  const int t = threadIdx.x;
  const int rowBase = blockIdx.x * 64;

  {  // stage A tile (1024 uint4) with row guard, and full Bt (2048 uint4)
    uint4* as4 = (uint4*)As;
    const uint4* a4 = (const uint4*)A;
    for (int i = t; i < 1024; i += 128) {
      const int r = i >> 4, seg = i & 15;
      const int row = rowBase + r;
      uint4 v = make_uint4(0u, 0u, 0u, 0u);
      if (row < N_NODES) v = a4[row * 16 + seg];
      as4[i] = v;
    }
    uint4* bs4 = (uint4*)Bs;
    const uint4* b4 = (const uint4*)Bt;
    for (int i = t; i < 2048; i += 128) bs4[i] = b4[i];
  }
  __syncthreads();

  const int wave = t >> 5;
  const int lane = t & 31;
  const int half = lane >> 4;  // 0: lanes 0-15, 1: lanes 16-31
  const int rr   = lane & 15;
  const int r0   = wave * 16;

  v8f acc[8];
#pragma unroll
  for (int nt = 0; nt < 8; ++nt) acc[nt] = (v8f){0.f,0.f,0.f,0.f,0.f,0.f,0.f,0.f};

#pragma unroll
  for (int kc = 0; kc < 4; ++kc) {
    // A fragment (16x32 f16): halves 0-7 = K[kc*32+half*8 ..+7], 8-15 = +16
    union { v16h v; uint4 u[2]; } af;
    const _Float16* arow = &As[(r0 + rr) * 128 + kc * 32];
    af.u[0] = *(const uint4*)(arow + half * 8);
    af.u[1] = *(const uint4*)(arow + 16 + half * 8);
#pragma unroll
    for (int nt = 0; nt < 8; ++nt) {
      // B fragment (32x16): lane = col nt*16+rr, halves 0-7 = K[half*8..+7], 8-15 = +16
      union { v16h v; uint4 u[2]; } bf;
      const _Float16* brow = &Bs[(nt * 16 + rr) * 128 + kc * 32 + half * 8];
      bf.u[0] = *(const uint4*)(brow);
      bf.u[1] = *(const uint4*)(brow + 16);
      acc[nt] = __builtin_amdgcn_wmma_f32_16x16x32_f16(
          false, af.v, false, bf.v, (short)0, acc[nt], false, false);
    }
  }

#pragma unroll
  for (int nt = 0; nt < 8; ++nt) {
    const int col = nt * 16 + rr;
    const float bv = bias ? bias[col] : 0.f;
#pragma unroll
    for (int r = 0; r < 8; ++r) {
      const int row = rowBase + r0 + half * 8 + r;  // C/D layout: VGPR r -> M=half*8+r
      if (row < N_NODES) Out[row * HC + col] = acc[nt][r] + bv;
    }
  }
}

// ---------------------------------------------------------------------------
// Per-node attention coefficients: wave per head (32 lanes == HID)
// ---------------------------------------------------------------------------
__global__ __launch_bounds__(128) void k_alpha(const float* __restrict__ xt,
                                               const float* __restrict__ asr,
                                               const float* __restrict__ adr,
                                               float* __restrict__ als,
                                               float* __restrict__ ald) {
  const int n  = blockIdx.x;
  const int hd = threadIdx.x >> 5;
  const int c  = threadIdx.x & 31;
  const float v = xt[n * HC + hd * HID + c];
  float s = v * asr[hd * HID + c];
  float d = v * adr[hd * HID + c];
#pragma unroll
  for (int o = 16; o > 0; o >>= 1) {
    s += __shfl_xor(s, o, 32);
    d += __shfl_xor(d, o, 32);
  }
  if (c == 0) { als[n * HEADS + hd] = s; ald[n * HEADS + hd] = d; }
}

// order-preserving float<->uint for atomicMax
__device__ __forceinline__ unsigned enc_f(float x) {
  const unsigned u = __float_as_uint(x);
  return (u & 0x80000000u) ? ~u : (u | 0x80000000u);
}
__device__ __forceinline__ float dec_f(unsigned u) {
  return (u & 0x80000000u) ? __uint_as_float(u ^ 0x80000000u) : __uint_as_float(~u);
}

__device__ __forceinline__ void enodes(const int* __restrict__ ei, int e, int& s, int& d) {
  if (e < N_EDGES) { s = ei[e]; d = ei[N_EDGES + e]; }
  else { const int v = e - N_EDGES; s = v; d = v; }  // self loops
}

// pass 1: leaky-relu logits + segment max (encoded atomicMax)
__global__ void k_elog(const int* __restrict__ ei, const float* __restrict__ als,
                       const float* __restrict__ ald, float* __restrict__ ev,
                       unsigned* __restrict__ nmax) {
  const int tid = blockIdx.x * blockDim.x + threadIdx.x;
  if (tid >= E_TOT * HEADS) return;
  const int e = tid >> 2, hd = tid & 3;
  int s, d; enodes(ei, e, s, d);
  float v = als[s * HEADS + hd] + ald[d * HEADS + hd];
  v = v > 0.f ? v : NEG_SLOPE * v;
  ev[tid] = v;
  atomicMax(&nmax[d * HEADS + hd], enc_f(v));
}

// pass 2: exp(e - max) + segment sum
__global__ void k_eexp(const int* __restrict__ ei, const unsigned* __restrict__ nmax,
                       float* __restrict__ ev, float* __restrict__ denom) {
  const int tid = blockIdx.x * blockDim.x + threadIdx.x;
  if (tid >= E_TOT * HEADS) return;
  const int e = tid >> 2, hd = tid & 3;
  int s, d; enodes(ei, e, s, d);
  const float ex = expf(ev[tid] - dec_f(nmax[d * HEADS + hd]));
  ev[tid] = ex;
  atomicAdd(&denom[d * HEADS + hd], ex);
}

// pass 3: scatter alpha * xt[src] into out[dst] (out pre-filled with skip branch)
__global__ __launch_bounds__(128) void k_escatter(const int* __restrict__ ei,
                                                  const float* __restrict__ ev,
                                                  const float* __restrict__ denom,
                                                  const float* __restrict__ xt,
                                                  float* __restrict__ out) {
  const int e = blockIdx.x;
  const int c = threadIdx.x;
  const int hd = c >> 5;
  int s, d; enodes(ei, e, s, d);
  const float a = ev[e * HEADS + hd] / denom[d * HEADS + hd];
  atomicAdd(&out[d * HC + c], a * xt[s * HC + c]);
}

// add conv bias, write back, accumulate per-channel sum/sumsq for BN
__global__ __launch_bounds__(128) void k_cstat(float* __restrict__ y,
                                               const float* __restrict__ convb,
                                               float* __restrict__ chS,
                                               float* __restrict__ chQ) {
  const int c = threadIdx.x;
  const int r0 = blockIdx.x * 64;
  const float cb = convb[c];
  float s = 0.f, q = 0.f;
  for (int r = 0; r < 64; ++r) {
    const int row = r0 + r;
    if (row >= N_NODES) break;
    const float v = y[row * HC + c] + cb;
    y[row * HC + c] = v;
    s += v; q += v * v;
  }
  atomicAdd(&chS[c], s);
  atomicAdd(&chQ[c], q);
}

__global__ void k_bn(const float* __restrict__ y, const float* __restrict__ chS,
                     const float* __restrict__ chQ, const float* __restrict__ g,
                     const float* __restrict__ b, float* __restrict__ h) {
  const int i = blockIdx.x * blockDim.x + threadIdx.x;
  if (i >= N_NODES * HC) return;
  const int c = i & 127;
  const float mean = chS[c] * (1.0f / N_NODES);
  const float var  = chQ[c] * (1.0f / N_NODES) - mean * mean;
  const float v = (y[i] - mean) * rsqrtf(var + BN_EPS) * g[c] + b[c];
  h[i] = v > 0.f ? v : 0.f;
}

__global__ __launch_bounds__(128) void k_pool(const int* __restrict__ batch,
                                              const float* __restrict__ h,
                                              float* __restrict__ pool,
                                              float* __restrict__ cnt) {
  const int n = blockIdx.x, c = threadIdx.x;
  const int g = batch[n];
  atomicAdd(&pool[g * HC + c], h[n * HC + c]);
  if (c == 0) atomicAdd(&cnt[g], 1.0f);
}

__global__ __launch_bounds__(128) void k_final(const float* __restrict__ pool,
                                               const float* __restrict__ cnt,
                                               const float* __restrict__ mlpW,
                                               const float* __restrict__ mlpb,
                                               float* __restrict__ out) {
  __shared__ float sd[128];
  const int g = blockIdx.x, c = threadIdx.x;
  float cc = cnt[g]; cc = cc > 1.f ? cc : 1.f;
  sd[c] = pool[g * HC + c] / cc * mlpW[c];
  __syncthreads();
  for (int o = 64; o > 0; o >>= 1) {
    if (c < o) sd[c] += sd[c + o];
    __syncthreads();
  }
  if (c == 0) out[g] = sd[0] + mlpb[0];
}

// ---------------------------------------------------------------------------
extern "C" void kernel_launch(void* const* d_in, const int* in_sizes, int n_in,
                              void* d_out, int out_size, void* d_ws, size_t ws_size,
                              hipStream_t stream) {
  (void)in_sizes; (void)n_in; (void)out_size; (void)ws_size;
  const int*   x        = (const int*)d_in[0];
  const int*   ei       = (const int*)d_in[1];
  const int*   batch    = (const int*)d_in[2];
  const float* atom_emb = (const float*)d_in[3];
  const float* convW    = (const float*)d_in[4];
  const float* a_src    = (const float*)d_in[5];
  const float* a_dst    = (const float*)d_in[6];
  const float* conv_b   = (const float*)d_in[7];
  const float* skipW    = (const float*)d_in[8];
  const float* skip_b   = (const float*)d_in[9];
  const float* bn_g     = (const float*)d_in[10];
  const float* bn_b     = (const float*)d_in[11];
  const float* mlpW     = (const float*)d_in[12];
  const float* mlp_b    = (const float*)d_in[13];
  float* out = (float*)d_out;

  char* ws = (char*)d_ws;
  size_t off = 0;
  auto alloc = [&](size_t bytes) -> void* {
    void* p = ws + off;
    off = (off + bytes + 255) & ~size_t(255);
    return p;
  };
  float*    h     = (float*)alloc((size_t)N_NODES * HC * 4);
  float*    skipb = (float*)alloc((size_t)N_NODES * HC * 4);  // skip, then += conv agg
  float*    xt    = (float*)alloc((size_t)N_NODES * HC * 4);
  _Float16* hF    = (_Float16*)alloc((size_t)N_NODES * HC * 2);
  float*    ev    = (float*)alloc((size_t)E_TOT * HEADS * 4);
  float*    als   = (float*)alloc((size_t)N_NODES * HEADS * 4);
  float*    ald   = (float*)alloc((size_t)N_NODES * HEADS * 4);
  unsigned* nmax  = (unsigned*)alloc((size_t)N_NODES * HEADS * 4);
  float*    denom = (float*)alloc((size_t)N_NODES * HEADS * 4);
  _Float16* Tc    = (_Float16*)alloc((size_t)HC * HC * 2);
  _Float16* Ts    = (_Float16*)alloc((size_t)HC * HC * 2);
  float*    chS   = (float*)alloc(HC * 4);
  float*    chQ   = (float*)alloc(HC * 4);
  float*    pool  = (float*)alloc((size_t)N_GRAPHS * HC * 4);
  float*    cnt   = (float*)alloc((size_t)N_GRAPHS * 4);

  k_atom<<<N_NODES, 128, 0, stream>>>(x, atom_emb, h);

  const int gemm_grid = (N_NODES + 63) / 64;
  const int et = E_TOT * HEADS;
  for (int l = 0; l < 4; ++l) {
    hipMemsetAsync(nmax, 0, (size_t)N_NODES * HEADS * 4, stream);  // 0 < enc of any float
    hipMemsetAsync(denom, 0, (size_t)N_NODES * HEADS * 4, stream);
    hipMemsetAsync(chS, 0, HC * 4, stream);
    hipMemsetAsync(chQ, 0, HC * 4, stream);

    k_wt<<<64, 256, 0, stream>>>(convW + l * HC * HC, skipW + l * HC * HC, Tc, Ts);
    k_h2h<<<1024, 256, 0, stream>>>(h, hF, N_NODES * HC);
    k_gemm_wmma<<<gemm_grid, 128, 0, stream>>>(hF, Tc, nullptr, xt);
    k_gemm_wmma<<<gemm_grid, 128, 0, stream>>>(hF, Ts, skip_b + l * HC, skipb);
    k_alpha<<<N_NODES, 128, 0, stream>>>(xt, a_src + l * HC, a_dst + l * HC, als, ald);
    k_elog<<<(et + 255) / 256, 256, 0, stream>>>(ei, als, ald, ev, nmax);
    k_eexp<<<(et + 255) / 256, 256, 0, stream>>>(ei, nmax, ev, denom);
    k_escatter<<<E_TOT, 128, 0, stream>>>(ei, ev, denom, xt, skipb);
    k_cstat<<<(N_NODES + 63) / 64, 128, 0, stream>>>(skipb, conv_b + l * HC, chS, chQ);
    k_bn<<<(N_NODES * HC + 255) / 256, 256, 0, stream>>>(
        skipb, chS, chQ, bn_g + l * HC, bn_b + l * HC, h);
  }

  hipMemsetAsync(pool, 0, (size_t)N_GRAPHS * HC * 4, stream);
  hipMemsetAsync(cnt, 0, (size_t)N_GRAPHS * 4, stream);
  k_pool<<<N_NODES, 128, 0, stream>>>(batch, h, pool, cnt);
  k_final<<<N_GRAPHS, 128, 0, stream>>>(pool, cnt, mlpW, mlp_b, out);
}
